// InterpretableMultiHeadAttention_47966194762431
// MI455X (gfx1250) — matrix operand order repaired and checked by reading, compiled, not run on hardware
//
#include <hip/hip_runtime.h>
#include <hip/hip_bf16.h>
#include <math.h>

// Problem constants (from reference)
#define NHEAD 16
#define DMODEL 1024
#define DK 64
#define BB 8
#define SS 1024
#define BS (BB * SS)   // 8192 rows total

typedef __attribute__((ext_vector_type(16))) _Float16 v16h;
typedef __attribute__((ext_vector_type(8)))  _Float16 v8h;
typedef __attribute__((ext_vector_type(4)))  _Float16 v4h;
typedef __attribute__((ext_vector_type(8)))  float    v8f;
typedef __attribute__((ext_vector_type(4)))  float    v4f;

// ---------------------------------------------------------------------------
// WMMA helper: D = A(16x32 f16) * B(32x16 f16) + C(16x16 f32)
// ---------------------------------------------------------------------------
__device__ __forceinline__ v8f wmma_f16(v16h a, v16h b, v8f c) {
  return __builtin_amdgcn_wmma_f32_16x16x32_f16(
      /*neg_a=*/false, a, /*neg_b=*/false, b,
      /*c_mod=*/(short)0, c, /*reuse_a=*/false, /*reuse_b=*/false);
}

// A-fragment (16x32, f16 source, row-major, leading dim ld in elements).
// ISA layout: lanes 0-15 row M=lane hold K=[0..7],[16..23]; lanes 16-31 hold
// K=[8..15],[24..31] -> two contiguous 16B runs per lane.
__device__ __forceinline__ v16h load_a_f16(const _Float16* base, int ld,
                                           int k0, int lane) {
  int r = lane & 15, hf = lane >> 4;
  const _Float16* p = base + (size_t)r * ld + k0 + 8 * hf;
  v16h a;
  v8h lo = *(const v8h*)(p);
  v8h hi = *(const v8h*)(p + 16);
#pragma unroll
  for (int i = 0; i < 8; ++i) { a[i] = lo[i]; a[8 + i] = hi[i]; }
  return a;
}

// A-fragment from f32 source (pooled accumulator), convert to f16.
__device__ __forceinline__ v16h load_a_f32cvt(const float* base, int ld,
                                              int k0, int lane) {
  int r = lane & 15, hf = lane >> 4;
  const float* p0 = base + (size_t)r * ld + k0 + 8 * hf;
  const float* p1 = p0 + 16;
  v16h a;
#pragma unroll
  for (int i = 0; i < 8; ++i) {
    a[i]     = (_Float16)p0[i];
    a[8 + i] = (_Float16)p1[i];
  }
  return a;
}

// B-fragment (32x16). Source is stored "transposed": row n (output column)
// holds its K-run contiguously with leading dim ld. Lane n = lane&15,
// lanes 0-15 hold K=[0..7],[16..23]; lanes 16-31 hold K=[8..15],[24..31].
__device__ __forceinline__ v16h load_b_f16(const _Float16* base, int ld,
                                           int k0, int lane) {
  int n = lane & 15, hf = lane >> 4;
  const _Float16* p = base + (size_t)n * ld + k0 + 8 * hf;
  v8h lo = *(const v8h*)(p);
  v8h hi = *(const v8h*)(p + 16);
  v16h b;
#pragma unroll
  for (int i = 0; i < 8; ++i) { b[i] = lo[i]; b[8 + i] = hi[i]; }
  return b;
}

// C/D 16x16 f32: VGPR r -> row r + 8*(lane>=16), col = lane&15.
__device__ __forceinline__ void store_c_f16(_Float16* p, int ld, v8f c,
                                            int lane) {
  int n = lane & 15, hf = lane >> 4;
#pragma unroll
  for (int r = 0; r < 8; ++r)
    p[(size_t)(r + 8 * hf) * ld + n] = (_Float16)c[r];
}

// ---------------------------------------------------------------------------
// Elementwise conversion / transpose-conversion / zero kernels
// ---------------------------------------------------------------------------
__global__ void convert_f32_to_f16(const float* __restrict__ in,
                                   _Float16* __restrict__ out, size_t n) {
  for (size_t i = (size_t)blockIdx.x * blockDim.x + threadIdx.x; i < n;
       i += (size_t)gridDim.x * blockDim.x)
    out[i] = (_Float16)in[i];
}

// in: [batch][R][C] f32 -> out: [batch][C][R] f16  (C = 1<<cshift)
__global__ void transpose_cvt(const float* __restrict__ in,
                              _Float16* __restrict__ out, int R, int cshift) {
  int C = 1 << cshift;
  size_t off = (size_t)blockIdx.y * R * C;
  int n = R * C;
  for (int idx = blockIdx.x * blockDim.x + threadIdx.x; idx < n;
       idx += gridDim.x * blockDim.x) {
    int r = idx >> cshift, c = idx & (C - 1);
    out[off + (size_t)c * R + r] = (_Float16)in[off + idx];
  }
}

__global__ void zero_f32(float* __restrict__ p, size_t n) {
  for (size_t i = (size_t)blockIdx.x * blockDim.x + threadIdx.x; i < n;
       i += (size_t)gridDim.x * blockDim.x)
    p[i] = 0.0f;
}

// ---------------------------------------------------------------------------
// Projection: qs[h] = q @ Wq[h], kk[h] = q @ Wk[h]  (blockIdx.y < NHEAD)
//             vsT    = (q @ Wv)^T per batch          (blockIdx.y == NHEAD)
// grid: (BS/16, NHEAD+1), block: 128 (4 waves, one N-tile of 16 each)
// ---------------------------------------------------------------------------
__global__ __launch_bounds__(128) void proj_kernel(
    const _Float16* __restrict__ q16, const _Float16* __restrict__ WqT,
    const _Float16* __restrict__ WkT, const _Float16* __restrict__ WvT,
    _Float16* __restrict__ qs, _Float16* __restrict__ kk,
    _Float16* __restrict__ vsT) {
  int m0 = blockIdx.x * 16;
  int y = blockIdx.y;
  int wave = threadIdx.x >> 5, lane = threadIdx.x & 31;
  int nb = wave * 16;  // column base within DK
  const _Float16* qrow = q16 + (size_t)m0 * DMODEL;

  if (y < NHEAD) {
    const _Float16* wq = WqT + (size_t)y * DK * DMODEL + (size_t)nb * DMODEL;
    const _Float16* wk = WkT + (size_t)y * DK * DMODEL + (size_t)nb * DMODEL;
    v8f cq = {}, ck = {};
#pragma unroll 4
    for (int kt = 0; kt < DMODEL / 32; ++kt) {
      v16h a  = load_a_f16(qrow, DMODEL, 32 * kt, lane);
      v16h bq = load_b_f16(wq, DMODEL, 32 * kt, lane);
      v16h bk = load_b_f16(wk, DMODEL, 32 * kt, lane);
      cq = wmma_f16(a, bq, cq);
      ck = wmma_f16(a, bk, ck);
    }
    store_c_f16(qs + ((size_t)y * BS + m0) * DK + nb, DK, cq, lane);
    store_c_f16(kk + ((size_t)y * BS + m0) * DK + nb, DK, ck, lane);
  } else {
    const _Float16* wv = WvT + (size_t)nb * DMODEL;
    v8f cv = {};
#pragma unroll 4
    for (int kt = 0; kt < DMODEL / 32; ++kt) {
      v16h a  = load_a_f16(qrow, DMODEL, 32 * kt, lane);
      v16h bv = load_b_f16(wv, DMODEL, 32 * kt, lane);
      cv = wmma_f16(a, bv, cv);
    }
    // store transposed: vsT[b][n][s]
    int b = m0 >> 10, s0 = m0 & (SS - 1);
    int n = lane & 15, hf = lane >> 4;
#pragma unroll
    for (int r = 0; r < 8; ++r)
      vsT[((size_t)b * DK + nb + n) * SS + s0 + r + 8 * hf] = (_Float16)cv[r];
  }
}

// ---------------------------------------------------------------------------
// Attention core: scores -> softmax -> attn (NT stores to d_out) ->
//                 head @ vs (f16 prob tile in LDS) -> pooled (atomic mean)
// grid: (BS/16, NHEAD), block: 128 (4 waves). 96KB LDS.
// ---------------------------------------------------------------------------
__global__ __launch_bounds__(128) void attn_kernel(
    const _Float16* __restrict__ qs, const _Float16* __restrict__ kk,
    const _Float16* __restrict__ vsT, float* __restrict__ attn_out,
    float* __restrict__ pooled) {
  __shared__ float    sc[16][SS];  // raw scores / exp values, 64 KB
  __shared__ _Float16 pf[16][SS];  // normalized probs (f16), 32 KB

  int m0 = blockIdx.x * 16;
  int h = blockIdx.y;
  int b = m0 >> 10, s0 = m0 & (SS - 1);
  int wave = threadIdx.x >> 5, lane = threadIdx.x & 31;
  const float scale = 0.125f;  // 1/sqrt(64)

  // A-fragments for the 16x64 qs tile (K = 64 -> two frags), shared by waves.
  const _Float16* qsrow = qs + ((size_t)h * BS + m0) * DK;
  v16h a0 = load_a_f16(qsrow, DK, 0, lane);
  v16h a1 = load_a_f16(qsrow, DK, 32, lane);

  // scores: wave w handles key tiles jt = w, w+4, ...
  const _Float16* kkbase = kk + ((size_t)h * BS + (size_t)b * SS) * DK;
  for (int jt = wave; jt < SS / 16; jt += 4) {
    const _Float16* kb = kkbase + (size_t)jt * 16 * DK;
    if (jt + 4 < SS / 16)
      __builtin_prefetch(kkbase + (size_t)(jt + 4) * 16 * DK, 0, 1);
    v16h b0 = load_b_f16(kb, DK, 0, lane);
    v16h b1 = load_b_f16(kb, DK, 32, lane);
    v8f c = {};
    c = wmma_f16(a0, b0, c);
    c = wmma_f16(a1, b1, c);
    int n = lane & 15, hf = lane >> 4;
#pragma unroll
    for (int r = 0; r < 8; ++r) sc[r + 8 * hf][jt * 16 + n] = c[r] * scale;
  }
  __syncthreads();

  // softmax: wave w owns rows 4w..4w+3, 8 lanes per row
  int rloc = lane >> 3;  // 0..3
  int row = wave * 4 + rloc;
  int cb = lane & 7;
  float mx = -1e30f;
#pragma unroll 8
  for (int i = 0; i < SS / 8; ++i) mx = fmaxf(mx, sc[row][cb + 8 * i]);
  mx = fmaxf(mx, __shfl_xor(mx, 1));
  mx = fmaxf(mx, __shfl_xor(mx, 2));
  mx = fmaxf(mx, __shfl_xor(mx, 4));
  float sum = 0.0f;
#pragma unroll 8
  for (int i = 0; i < SS / 8; ++i) {
    float e = __expf(sc[row][cb + 8 * i] - mx);
    sc[row][cb + 8 * i] = e;
    sum += e;
  }
  sum += __shfl_xor(sum, 1);
  sum += __shfl_xor(sum, 2);
  sum += __shfl_xor(sum, 4);
  float inv = 1.0f / sum;

  // normalize: f32 probs stream to d_out (non-temporal, never re-read),
  // f16 probs land in LDS for the attn@V WMMA phase.
#pragma unroll
  for (int rr = 0; rr < 4; ++rr) {
    int rrow = wave * 4 + rr;
    float vinv = __shfl(inv, rr * 8);
    float* arow =
        attn_out + (((size_t)h * BB + b) * SS + (s0 + rrow)) * (size_t)SS;
    for (int c4 = lane; c4 < SS / 4; c4 += 32) {
      v4f pv = *(const v4f*)&sc[rrow][c4 * 4];
      pv *= vinv;
      __builtin_nontemporal_store(pv, (v4f*)(arow + c4 * 4));
      v4h ph;
#pragma unroll
      for (int i = 0; i < 4; ++i) ph[i] = (_Float16)pv[i];
      *(v4h*)&pf[rrow][c4 * 4] = ph;
    }
  }
  __syncthreads();

  // head = attn @ vs : wave w computes N-tile w of the 16x64 output
  int nb = wave * 16;
  const _Float16* vb = vsT + ((size_t)b * DK + nb) * SS;
  v8f cacc = {};
#pragma unroll 4
  for (int kt = 0; kt < SS / 32; ++kt) {
    v16h a  = load_a_f16(&pf[0][0], SS, 32 * kt, lane);
    v16h bv = load_b_f16(vb, SS, 32 * kt, lane);
    cacc = wmma_f16(a, bv, cacc);
  }
  // mean over heads: atomically accumulate head/NHEAD
  int n = lane & 15, hf = lane >> 4;
#pragma unroll
  for (int r = 0; r < 8; ++r)
    atomicAdd(&pooled[(size_t)(m0 + r + 8 * hf) * DK + nb + n],
              cacc[r] * (1.0f / NHEAD));
}

// ---------------------------------------------------------------------------
// Output projection: out = pooled @ Wo
// grid: (BS/16, DMODEL/64), block: 128 (4 waves x 16-col N-tiles)
// ---------------------------------------------------------------------------
__global__ __launch_bounds__(128) void out_kernel(
    const float* __restrict__ pooled, const _Float16* __restrict__ WoT,
    float* __restrict__ out) {
  int m0 = blockIdx.x * 16;
  int wave = threadIdx.x >> 5, lane = threadIdx.x & 31;
  int nb = (blockIdx.y * 4 + wave) * 16;  // column base in DMODEL
  const float* prow = pooled + (size_t)m0 * DK;
  const _Float16* wb = WoT + (size_t)nb * DK;
  v8f c = {};
#pragma unroll
  for (int kt = 0; kt < DK / 32; ++kt) {
    v16h a  = load_a_f32cvt(prow, DK, 32 * kt, lane);
    v16h bo = load_b_f16(wb, DK, 32 * kt, lane);
    c = wmma_f16(a, bo, c);
  }
  int n = lane & 15, hf = lane >> 4;
#pragma unroll
  for (int r = 0; r < 8; ++r)
    __builtin_nontemporal_store(
        c[r], out + (size_t)(m0 + r + 8 * hf) * DMODEL + nb + n);
}

// ---------------------------------------------------------------------------
// Host launcher
// ---------------------------------------------------------------------------
extern "C" void kernel_launch(void* const* d_in, const int* in_sizes, int n_in,
                              void* d_out, int out_size, void* d_ws,
                              size_t ws_size, hipStream_t stream) {
  const float* q  = (const float*)d_in[0];
  // d_in[1] (k), d_in[2] (v) are unused by the reference
  const float* Wq = (const float*)d_in[3];
  const float* Wk = (const float*)d_in[4];
  const float* Wv = (const float*)d_in[5];
  const float* Wo = (const float*)d_in[6];

  float* out_p  = (float*)d_out;                     // [B,S,D]
  float* attn_p = out_p + (size_t)BB * SS * DMODEL;  // [H,B,S,S]

  // Workspace carve-up (256B aligned)
  char* ws = (char*)d_ws;
  auto carve = [&](size_t bytes) {
    void* p = (void*)ws;
    ws += (bytes + 255) & ~(size_t)255;
    return p;
  };
  _Float16* q16  = (_Float16*)carve((size_t)BS * DMODEL * 2);
  _Float16* WqT  = (_Float16*)carve((size_t)NHEAD * DK * DMODEL * 2);
  _Float16* WkT  = (_Float16*)carve((size_t)NHEAD * DK * DMODEL * 2);
  _Float16* WvT  = (_Float16*)carve((size_t)DK * DMODEL * 2);
  _Float16* WoT  = (_Float16*)carve((size_t)DMODEL * DK * 2);
  _Float16* qs16 = (_Float16*)carve((size_t)NHEAD * BS * DK * 2);
  _Float16* kk16 = (_Float16*)carve((size_t)NHEAD * BS * DK * 2);
  _Float16* vsT  = (_Float16*)carve((size_t)BB * DK * SS * 2);
  float*    pooled = (float*)carve((size_t)BS * DK * 4);

  // Phase 0: precision conversion + weight transposes (C is a power of two)
  convert_f32_to_f16<<<2048, 256, 0, stream>>>(q, q16, (size_t)BS * DMODEL);
  transpose_cvt<<<dim3(64, NHEAD), 256, 0, stream>>>(Wq, WqT, DMODEL, 6);
  transpose_cvt<<<dim3(64, NHEAD), 256, 0, stream>>>(Wk, WkT, DMODEL, 6);
  transpose_cvt<<<dim3(64, 1), 256, 0, stream>>>(Wv, WvT, DMODEL, 6);
  transpose_cvt<<<dim3(64, 1), 256, 0, stream>>>(Wo, WoT, DK, 10);
  zero_f32<<<512, 256, 0, stream>>>(pooled, (size_t)BS * DK);

  // Phase 1: projections (qs, kk per head; vs shared, stored transposed)
  proj_kernel<<<dim3(BS / 16, NHEAD + 1), 128, 0, stream>>>(
      q16, WqT, WkT, WvT, qs16, kk16, vsT);

  // Phase 2: scores + softmax + attn write + head@vs + head-mean pooling
  attn_kernel<<<dim3(BS / 16, NHEAD), 128, 0, stream>>>(qs16, kk16, vsT,
                                                        attn_p, pooled);

  // Phase 3: output projection
  out_kernel<<<dim3(BS / 16, DMODEL / 64), 128, 0, stream>>>(pooled, WoT,
                                                             out_p);
}
